// BidirectionalRNN_69389491634338
// MI455X (gfx1250) — matrix operand order repaired and checked by reading, not compiled, not run on hardware
//
#include <hip/hip_runtime.h>
#include <stdint.h>

// Problem constants (match reference)
#define N_ 256
#define T_ 128
#define D_ 1024
#define H_ 1024
#define O_ 1024

#define NBLK_RNN 64u   // persistent grid (co-resident: 64KB LDS, 8 waves each)

typedef __bf16 bf16_t;
typedef __attribute__((ext_vector_type(16))) __bf16 v16bf;
typedef __attribute__((ext_vector_type(8)))  __bf16 v8bf;
typedef __attribute__((ext_vector_type(8)))  float  v8f;
typedef __attribute__((ext_vector_type(4)))  unsigned int u32x4;
typedef __attribute__((ext_vector_type(8)))  int          i32x8;
typedef __attribute__((ext_vector_type(4)))  int          i32x4;

#if __has_builtin(__builtin_amdgcn_tensor_load_to_lds)
#define HAVE_TDM 1
#endif

// ---------- bf16 <-> f32 helpers (round-to-nearest-even) ----------
__device__ __forceinline__ bf16_t f2bf(float f) {
    unsigned u = __builtin_bit_cast(unsigned, f);
    unsigned r = (u + 0x7fffu + ((u >> 16) & 1u)) >> 16;
    return __builtin_bit_cast(bf16_t, (unsigned short)r);
}
__device__ __forceinline__ float bf2f(bf16_t h) {
    unsigned short s = __builtin_bit_cast(unsigned short, h);
    return __builtin_bit_cast(float, ((unsigned)s) << 16);
}

// ---------- WMMA fragment helpers (contiguous k order, same for A and B) ----------
// A (16x32 bf16) from an LDS-staged row-major strip (ld = 1024): lane l holds
// row rowoff+(l&15); elems j=0..15 -> K=hi*16+j.
__device__ __forceinline__ v16bf load_a_lds(const bf16_t* aS, int rowoff, int k0,
                                            int lo16, int hi) {
    const bf16_t* p = aS + (rowoff + lo16) * 1024 + k0 + hi * 16;
    v8bf lo = *(const v8bf*)p;
    v8bf hh = *(const v8bf*)(p + 8);
    return __builtin_shufflevector(lo, hh, 0,1,2,3,4,5,6,7,8,9,10,11,12,13,14,15);
}
// B (32x16 bf16) from pre-transposed weights WT[n][k]: lane l holds column
// (l&15); elems j -> K = hi*16 + j.  One 32B contiguous load per lane.
__device__ __forceinline__ v16bf load_b_frag(const bf16_t* colbase, int k0, int hi) {
    return *(const v16bf*)(colbase + k0 + hi * 16);
}
__device__ __forceinline__ v8f wmma_bf16(v16bf a, v16bf b, v8f c) {
    return __builtin_amdgcn_wmma_f32_16x16x32_bf16(false, a, false, b, (short)0, c,
                                                   false, false);
}

// ---------- stage a 32-row x 1024-col bf16 A strip into LDS (64KB) ----------
// Tensor Data Mover: one 2D descriptor (tile 1024x32, elem 2B, row stride
// row_stride_elems), issued by wave 0, completion via TENSORcnt.
__device__ __forceinline__ void stage_a_tile(const bf16_t* gsrc, bf16_t* lds_dst,
                                             unsigned row_stride_elems) {
#ifdef HAVE_TDM
    if ((threadIdx.x >> 5) == 0) {
        unsigned long long ga = (unsigned long long)(uintptr_t)gsrc;
        u32x4 g0;
        g0[0] = 1u;                                   // count=1, user-mode descriptor
        g0[1] = (unsigned)(uintptr_t)lds_dst;         // lds_addr (addr[31:0])
        g0[2] = (unsigned)ga;                         // global_addr[31:0]
        g0[3] = (unsigned)((ga >> 32) & 0x01ffffffull) | 0x80000000u; // addr[56:32]|type=2
        i32x8 g1;
        g1[0] = (int)(1u << 16);                      // workgroup_mask=0, data_size=1 (2B)
        g1[1] = (int)(1024u << 16);                   // tensor_dim0 = 1024
        g1[2] = (int)(32u << 16);                     // tensor_dim1 = 32
        g1[3] = (int)(1024u << 16);                   // tile_dim0 = 1024
        g1[4] = 32;                                   // tile_dim1 = 32, tile_dim2 = 0
        g1[5] = (int)row_stride_elems;                // tensor_dim0_stride[31:0]
        g1[6] = 0;
        g1[7] = 0;
        i32x4 z4 = {0, 0, 0, 0};                      // groups 2/3 unused (2D tensor)
#if defined(__clang_major__) && __clang_major__ >= 23
        i32x8 z8 = {0, 0, 0, 0, 0, 0, 0, 0};
        __builtin_amdgcn_tensor_load_to_lds(g0, g1, z4, z4, z8, 0);
#else
        __builtin_amdgcn_tensor_load_to_lds(g0, g1, z4, z4, 0);
#endif
#if __has_builtin(__builtin_amdgcn_s_wait_tensorcnt)
        __builtin_amdgcn_s_wait_tensorcnt(0);
#else
        asm volatile("s_wait_tensorcnt 0x0" ::: "memory");
#endif
    }
    __syncthreads();
#else
    // Cooperative fallback: 256 threads, 8 per row, 16 x 16B chunks each.
    int r = threadIdx.x >> 3;
    int c = threadIdx.x & 7;
    const v8bf* src = (const v8bf*)(gsrc + (size_t)r * row_stride_elems);
    v8bf* dst = (v8bf*)(lds_dst + r * 1024);
#pragma unroll
    for (int i = 0; i < 16; ++i) dst[c + 8 * i] = src[c + 8 * i];
    __syncthreads();
#endif
}

// ---------- GEMM core: 32 rows (LDS) x 32 cols per wave, K = 1024 ------------
// 2 A frags x 2 B frags -> 4 WMMAs per K step (vmem-per-wmma halved vs 16x64).
// K unrolled by 2 so the two B register buffers alternate roles.
__device__ __forceinline__ void gemm_strip32(const bf16_t* aS, const bf16_t* bb,
                                             int lo16, int hi, v8f acc[4]) {
    v16bf b0[2], b1[2];
#pragma unroll
    for (int j = 0; j < 2; ++j) b0[j] = load_b_frag(bb + (size_t)j * 16 * 1024, 0, hi);
    for (int k0 = 0; k0 < 1024 - 64; k0 += 64) {
#pragma unroll
        for (int j = 0; j < 2; ++j)
            b1[j] = load_b_frag(bb + (size_t)j * 16 * 1024, k0 + 32, hi);
        {
            v16bf a0 = load_a_lds(aS, 0, k0, lo16, hi);
            v16bf a1 = load_a_lds(aS, 16, k0, lo16, hi);
            acc[0] = wmma_bf16(a0, b0[0], acc[0]);
            acc[1] = wmma_bf16(a0, b0[1], acc[1]);
            acc[2] = wmma_bf16(a1, b0[0], acc[2]);
            acc[3] = wmma_bf16(a1, b0[1], acc[3]);
        }
#pragma unroll
        for (int j = 0; j < 2; ++j)
            b0[j] = load_b_frag(bb + (size_t)j * 16 * 1024, k0 + 64, hi);
        {
            v16bf a0 = load_a_lds(aS, 0, k0 + 32, lo16, hi);
            v16bf a1 = load_a_lds(aS, 16, k0 + 32, lo16, hi);
            acc[0] = wmma_bf16(a0, b1[0], acc[0]);
            acc[1] = wmma_bf16(a0, b1[1], acc[1]);
            acc[2] = wmma_bf16(a1, b1[0], acc[2]);
            acc[3] = wmma_bf16(a1, b1[1], acc[3]);
        }
    }
    // tail pair: k0 = 960 / 992
#pragma unroll
    for (int j = 0; j < 2; ++j)
        b1[j] = load_b_frag(bb + (size_t)j * 16 * 1024, 992, hi);
    {
        v16bf a0 = load_a_lds(aS, 0, 960, lo16, hi);
        v16bf a1 = load_a_lds(aS, 16, 960, lo16, hi);
        acc[0] = wmma_bf16(a0, b0[0], acc[0]);
        acc[1] = wmma_bf16(a0, b0[1], acc[1]);
        acc[2] = wmma_bf16(a1, b0[0], acc[2]);
        acc[3] = wmma_bf16(a1, b0[1], acc[3]);
    }
    {
        v16bf a0 = load_a_lds(aS, 0, 992, lo16, hi);
        v16bf a1 = load_a_lds(aS, 16, 992, lo16, hi);
        acc[0] = wmma_bf16(a0, b1[0], acc[0]);
        acc[1] = wmma_bf16(a0, b1[1], acc[1]);
        acc[2] = wmma_bf16(a1, b1[0], acc[2]);
        acc[3] = wmma_bf16(a1, b1[1], acc[3]);
    }
}

// ---------- device-wide arrive-and-spin barrier (persistent kernel) ----------
__device__ __forceinline__ void grid_sync(unsigned* cnt, unsigned target) {
    __threadfence();
    __syncthreads();
    if (threadIdx.x == 0) {
        __hip_atomic_fetch_add(cnt, 1u, __ATOMIC_ACQ_REL, __HIP_MEMORY_SCOPE_AGENT);
        while (__hip_atomic_load(cnt, __ATOMIC_ACQUIRE, __HIP_MEMORY_SCOPE_AGENT) < target)
            __builtin_amdgcn_s_sleep(8);
    }
    __syncthreads();
}

// ---------- elementwise prep kernels ----------
__global__ __launch_bounds__(256) void conv_x_kernel(const float* __restrict__ x,
                                                     bf16_t* __restrict__ xb) {
    size_t i = ((size_t)blockIdx.x * blockDim.x + threadIdx.x) * 8;
#pragma unroll
    for (int j = 0; j < 8; ++j) xb[i + j] = f2bf(x[i + j]);
}

// All 5 weight matrices (1024x1024, row-major K-major) -> bf16 transposed
// (N-major).  4096 blocks per matrix; `which` is block-uniform.
__global__ __launch_bounds__(256) void conv_wT_all_kernel(
    const float* __restrict__ s0, const float* __restrict__ s1,
    const float* __restrict__ s2, const float* __restrict__ s3,
    const float* __restrict__ s4, bf16_t* __restrict__ d0, bf16_t* __restrict__ d1,
    bf16_t* __restrict__ d2, bf16_t* __restrict__ d3, bf16_t* __restrict__ d4) {
    int which = blockIdx.x >> 12;
    const float* W = which == 0 ? s0 : which == 1 ? s1 : which == 2 ? s2
                   : which == 3 ? s3 : s4;
    bf16_t* WT = which == 0 ? d0 : which == 1 ? d1 : which == 2 ? d2
               : which == 3 ? d3 : d4;
    int idx = (blockIdx.x & 4095) * 256 + threadIdx.x;
    int n = idx >> 10;
    int k = idx & 1023;
    WT[idx] = f2bf(W[(size_t)k * 1024 + n]);
}

__global__ __launch_bounds__(256) void init_kernel(bf16_t* __restrict__ hf0,
                                                   bf16_t* __restrict__ hb0,
                                                   unsigned* __restrict__ cnt) {
    int i = blockIdx.x * blockDim.x + threadIdx.x;
    if (i == 0) *cnt = 0u;
    bf16_t z = __builtin_bit_cast(bf16_t, (unsigned short)0);
    hf0[i] = z;
    hb0[i] = z;
}

// ---------- U = X @ Wxh (both directions; rev bakes in (-t)%T indexing) ------
// One block: 32 batch rows (TDM-staged) x 256 hidden cols (8 waves x 32).
// grid per dir = 4 colgrps * 8 batch tiles * T = 4096 blocks.
__global__ __launch_bounds__(256) void gemm_u_kernel(const bf16_t* __restrict__ xb,
                                                     const bf16_t* __restrict__ WT,
                                                     bf16_t* __restrict__ U, int rev) {
    __shared__ bf16_t aS[32 * 1024];
    const int lane = threadIdx.x & 31, wave = threadIdx.x >> 5;
    const int hi = lane >> 4, lo16 = lane & 15;
    int colgrp = blockIdx.x & 3;
    int rest = blockIdx.x >> 2;
    int nb = rest & 7;
    int t = rest >> 3;
    int tsrc = rev ? ((T_ - t) & (T_ - 1)) : t;  // (-t) % T  (t=0 -> 0)

    stage_a_tile(xb + ((size_t)(nb * 32) * T_ + tsrc) * D_, aS, T_ * D_);

    int col0 = colgrp * 256 + wave * 32;
    const bf16_t* bb = WT + (size_t)(col0 + lo16) * D_;
    v8f acc[4] = {};
    gemm_strip32(aS, bb, lo16, hi, acc);

#pragma unroll
    for (int mi = 0; mi < 2; ++mi)
#pragma unroll
        for (int nj = 0; nj < 2; ++nj) {
            int col = col0 + nj * 16 + lo16;
#pragma unroll
            for (int r = 0; r < 8; ++r) {
                size_t row = (size_t)t * N_ + nb * 32 + mi * 16 + r + hi * 8;
                U[row * H_ + col] = f2bf(acc[mi * 2 + nj][r]);
            }
        }
}

// ---------- persistent recurrence + fused output projection ------------------
// ONE launch: 64 blocks (dir x 8 row-tiles x 4 colgrps) loop over t = 0..T-1
// with a device-scope barrier per step; after the final barrier, dir==0 blocks
// compute out = (hf+hb) @ WhoT.
__global__ __launch_bounds__(256) void rnn_persist_kernel(
    bf16_t* __restrict__ hf0, bf16_t* __restrict__ hf1,
    bf16_t* __restrict__ hb0, bf16_t* __restrict__ hb1,
    const bf16_t* __restrict__ WT_f, const bf16_t* __restrict__ WT_b,
    const bf16_t* __restrict__ U_f, const bf16_t* __restrict__ U_b,
    const bf16_t* __restrict__ WhoT, float* __restrict__ out,
    unsigned* __restrict__ cnt) {
    __shared__ bf16_t aS[32 * 1024];
    const int lane = threadIdx.x & 31, wave = threadIdx.x >> 5;
    const int hi = lane >> 4, lo16 = lane & 15;
    int dir = blockIdx.x >> 5;
    int ww = blockIdx.x & 31;
    int colgrp = ww & 3;
    int nb = ww >> 2;

    bf16_t* hpp[2];
    hpp[0] = dir ? hb0 : hf0;
    hpp[1] = dir ? hb1 : hf1;
    const bf16_t* WT = dir ? WT_b : WT_f;
    const bf16_t* U = dir ? U_b : U_f;

    const int col0 = colgrp * 256 + wave * 32;
    const bf16_t* bb = WT + (size_t)(col0 + lo16) * H_;

    for (int t = 0; t < T_; ++t) {
        const bf16_t* hin = hpp[t & 1];
        bf16_t* hout = hpp[(t + 1) & 1];

        stage_a_tile(hin + (size_t)(nb * 32) * H_, aS, H_);
        v8f acc[4] = {};
        gemm_strip32(aS, bb, lo16, hi, acc);

#pragma unroll
        for (int mi = 0; mi < 2; ++mi)
#pragma unroll
            for (int nj = 0; nj < 2; ++nj) {
                int col = col0 + nj * 16 + lo16;
#pragma unroll
                for (int r = 0; r < 8; ++r) {
                    int row = nb * 32 + mi * 16 + r + hi * 8;
                    float v = acc[mi * 2 + nj][r] +
                              bf2f(U[((size_t)t * N_ + row) * H_ + col]);
                    hout[(size_t)row * H_ + col] = f2bf(v);
                }
            }
        grid_sync(cnt, NBLK_RNN * (unsigned)(t + 1));
    }

    // Final h is in hf0/hb0 (t=127 writes buffer (127+1)&1 = 0).
    if (dir == 0) {
        // Stage hsum tile (hf+hb, rows nb*32..+31) into LDS.
        int r = threadIdx.x >> 3;   // 8 threads per row
        int c = threadIdx.x & 7;
        const bf16_t* pf = hf0 + (size_t)(nb * 32 + r) * H_;
        const bf16_t* pb = hb0 + (size_t)(nb * 32 + r) * H_;
        bf16_t* dst = aS + r * 1024;
#pragma unroll
        for (int i = 0; i < 16; ++i) {
            int base = (c + 8 * i) * 16;
#pragma unroll
            for (int e = 0; e < 16; ++e)
                dst[base + e] = f2bf(bf2f(pf[base + e]) + bf2f(pb[base + e]));
        }
        __syncthreads();

        const bf16_t* bbo = WhoT + (size_t)(col0 + lo16) * H_;
        v8f acc[4] = {};
        gemm_strip32(aS, bbo, lo16, hi, acc);
#pragma unroll
        for (int mi = 0; mi < 2; ++mi)
#pragma unroll
            for (int nj = 0; nj < 2; ++nj) {
                int col = col0 + nj * 16 + lo16;
#pragma unroll
                for (int r2 = 0; r2 < 8; ++r2)
                    out[(size_t)(nb * 32 + mi * 16 + r2 + hi * 8) * O_ + col] =
                        acc[mi * 2 + nj][r2];
            }
    }
}

// ---------- host launcher ----------
extern "C" void kernel_launch(void* const* d_in, const int* in_sizes, int n_in,
                              void* d_out, int out_size, void* d_ws, size_t ws_size,
                              hipStream_t stream) {
    (void)in_sizes; (void)n_in; (void)out_size; (void)ws_size;
    const float* x = (const float*)d_in[0];
    const float* Wxh_f = (const float*)d_in[1];
    const float* Whh_f = (const float*)d_in[2];
    const float* Wxh_b = (const float*)d_in[3];
    const float* Whh_b = (const float*)d_in[4];
    const float* Who = (const float*)d_in[5];
    float* out = (float*)d_out;

    char* base = (char*)d_ws;
    size_t off = 0;
    auto alloc = [&](size_t bytes) -> void* {
        void* p = base + off;
        off += (bytes + 255) & ~(size_t)255;
        return p;
    };
    const size_t NTD = (size_t)N_ * T_ * D_;
    const size_t MAT = (size_t)1024 * 1024;
    bf16_t* xb     = (bf16_t*)alloc(NTD * sizeof(bf16_t));                  // 64 MiB
    bf16_t* WxhT_f = (bf16_t*)alloc(MAT * sizeof(bf16_t));
    bf16_t* WhhT_f = (bf16_t*)alloc(MAT * sizeof(bf16_t));
    bf16_t* WxhT_b = (bf16_t*)alloc(MAT * sizeof(bf16_t));
    bf16_t* WhhT_b = (bf16_t*)alloc(MAT * sizeof(bf16_t));
    bf16_t* WhoT   = (bf16_t*)alloc(MAT * sizeof(bf16_t));
    bf16_t* U_f    = (bf16_t*)alloc((size_t)T_ * N_ * H_ * sizeof(bf16_t)); // 64 MiB
    bf16_t* U_b    = (bf16_t*)alloc((size_t)T_ * N_ * H_ * sizeof(bf16_t)); // 64 MiB
    bf16_t* hbuf[4];
    for (int i = 0; i < 4; ++i) hbuf[i] = (bf16_t*)alloc((size_t)N_ * H_ * sizeof(bf16_t));
    unsigned* cnt = (unsigned*)alloc(256);

    // 1) precision conversion + weight transpose (bf16, N-major for B-fragments)
    conv_x_kernel<<<16384, 256, 0, stream>>>(x, xb);
    conv_wT_all_kernel<<<5 * 4096, 256, 0, stream>>>(
        Wxh_f, Whh_f, Wxh_b, Whh_b, Who, WxhT_f, WhhT_f, WxhT_b, WhhT_b, WhoT);
    init_kernel<<<1024, 256, 0, stream>>>(hbuf[0], hbuf[2], cnt);

    // 2) all input projections up front (parallel over T) — half the total FLOPs
    gemm_u_kernel<<<4096, 256, 0, stream>>>(xb, WxhT_f, U_f, 0);
    gemm_u_kernel<<<4096, 256, 0, stream>>>(xb, WxhT_b, U_b, 1);

    // 3) ONE persistent launch: 128 recurrence steps (device-wide barrier per
    //    step) + fused (hf+hb) @ Who -> f32 out.
    rnn_persist_kernel<<<NBLK_RNN, 256, 0, stream>>>(
        hbuf[0], hbuf[1], hbuf[2], hbuf[3], WhhT_f, WhhT_b, U_f, U_b, WhoT, out, cnt);
}
